// QuantLinear_74809740362014
// MI455X (gfx1250) — matrix operand order repaired and checked by reading, not compiled, not run on hardware
//
#include <hip/hip_runtime.h>
#include <stdint.h>

typedef __attribute__((ext_vector_type(16))) _Float16 v16h;
typedef __attribute__((ext_vector_type(8)))  _Float16 v8h;
typedef __attribute__((ext_vector_type(2)))  _Float16 v2h;
typedef __attribute__((ext_vector_type(8)))  float    v8f;

union AU { v16h v; v8h h8[2]; };
union BU { v16h v; v2h h2[8]; };
union PU { uint32_t u; v2h h; };

// Expand nibble pair j (bits 8j..8j+7) of packed word w into two fp16 values
// (q - zp) * s, computed exactly: (0x6400|n) == fp16(1024+n), then exact
// packed subtract of fp16(1024+zp) and packed multiply by scale.
static __device__ __forceinline__ v2h dq_pair(uint32_t w, int j, v2h z2, v2h s2) {
    uint32_t b = (w >> (8 * j)) & 0xFFu;
    PU p;
    p.u = (b & 0xFu) | ((b & 0xF0u) << 12) | 0x64006400u;
    return (p.h - z2) * s2;   // v_pk_add_f16(neg) + v_pk_mul_f16
}

// gfx1250 async global->LDS DMA (ASYNCcnt); per-lane 16B transfer.
static __device__ __forceinline__ void async_ld_b128(uint32_t lds_off, const void* gaddr) {
    asm volatile("global_load_async_to_lds_b128 %0, %1, off"
                 :: "v"(lds_off), "v"((uint64_t)(uintptr_t)gaddr)
                 : "memory");
}
static __device__ __forceinline__ void wait_async0() {
    asm volatile("s_wait_asynccnt 0x0" ::: "memory");
}

#define BM 128
#define BN 128
#define BK 32
#define LDS_STRIDE 40   // halves per row (80 bytes): conflict-free b128 reads

template <int CN, int CK>
__global__ __launch_bounds__(256)
void gptq4_wmma_gemm(const _Float16* __restrict__ x,
                     const int* __restrict__ qweight,
                     const int* __restrict__ qzeros,
                     const _Float16* __restrict__ scales,
                     float* __restrict__ out,
                     int Nrt, int Krt) {
    const int N = CN ? CN : Nrt;
    const int K = CK ? CK : Krt;

    __shared__ __align__(16) _Float16 ldsx[2][BM * LDS_STRIDE];

    const int tid    = threadIdx.x;
    const int wid    = tid >> 5;       // 8 waves
    const int lid    = tid & 31;
    const int lane16 = lid & 15;
    const int lhalf  = lid >> 4;       // 0: lanes 0-15, 1: lanes 16-31
    const int wm     = wid >> 2;       // 0..1  -> 64 rows each
    const int wn     = wid & 3;        // 0..3  -> 32 cols each
    const int mbase  = blockIdx.y * BM;
    const int nbase  = blockIdx.x * BN;

    // Per-lane output columns for the 2 N-tiles; dequant constants.
    int gcol[2];
    v2h s2[2], z2[2];
#pragma unroll
    for (int nt = 0; nt < 2; ++nt) {
        int c = nbase + wn * 32 + nt * 16 + lane16;
        gcol[nt] = c;
        int zq = ((qzeros[c >> 3] >> (4 * (c & 7))) & 0xF) + 1;   // zero + 1
        _Float16 s = scales[c];
        _Float16 zz = (_Float16)(float)(1024 + zq);               // exact in fp16
        s2[nt] = (v2h){s, s};
        z2[nt] = (v2h){zz, zz};
    }

    // x-tile staging: 128 rows x 64B = 512 chunks of 16B; 2 chunks per thread,
    // DMA'd straight into LDS via the async path.
    const int row0 = tid >> 2,         col0 = tid & 3;
    const int row1 = (tid + 256) >> 2, col1 = tid & 3;  // (tid+256)&3 == tid&3

    const _Float16* xp0 = x + (size_t)(mbase + row0) * K + col0 * 8;
    const _Float16* xp1 = x + (size_t)(mbase + row1) * K + col1 * 8;

    uint32_t la0[2], la1[2];
#pragma unroll
    for (int b = 0; b < 2; ++b) {
        la0[b] = (uint32_t)(uintptr_t)&ldsx[b][row0 * LDS_STRIDE + col0 * 8];
        la1[b] = (uint32_t)(uintptr_t)&ldsx[b][row1 * LDS_STRIDE + col1 * 8];
    }

    // Weight pointers: row pair (kb, kb+1) with kb = kk*4 + lhalf*2.
    const int* wp0 = qweight + (size_t)(lhalf * 2) * N + gcol[0];
    const int* wp1 = qweight + (size_t)(lhalf * 2) * N + gcol[1];

    v8f acc[4][2] = {};
    const int nk = K >> 5;   // K / 32

    // Prologue: DMA k-tile 0 into LDS buffer 0.
    async_ld_b128(la0[0], xp0);
    async_ld_b128(la1[0], xp1);
    wait_async0();
    __syncthreads();

#pragma unroll 2
    for (int kk = 0; kk < nk; ++kk) {
        const int buf = kk & 1;

        // Kick off DMA of the next x tile into the other buffer; it overlaps
        // the whole dequant + WMMA body below.
        if (kk + 1 < nk) {
            xp0 += BK; xp1 += BK;
            async_ld_b128(la0[buf ^ 1], xp0);
            async_ld_b128(la1[buf ^ 1], xp1);
        }

        // B fragments: lane covers K range [c0, c0+16), c0 = lhalf*16.
        uint32_t w00 = (uint32_t)wp0[0];
        uint32_t w01 = (uint32_t)wp0[N];     // immediate offset when CN != 0
        uint32_t w10 = (uint32_t)wp1[0];
        uint32_t w11 = (uint32_t)wp1[N];
        wp0 += 4 * (size_t)N; wp1 += 4 * (size_t)N;
        __builtin_prefetch(wp0, 0, 1);
        __builtin_prefetch(wp1, 0, 1);

        BU b[2];
#pragma unroll
        for (int j = 0; j < 4; ++j) {
            b[0].h2[j]     = dq_pair(w00, j, z2[0], s2[0]);
            b[0].h2[4 + j] = dq_pair(w01, j, z2[0], s2[0]);
            b[1].h2[j]     = dq_pair(w10, j, z2[1], s2[1]);
            b[1].h2[4 + j] = dq_pair(w11, j, z2[1], s2[1]);
        }

        // Hoist all A fragments (8 x ds_load_b128), then 8 back-to-back WMMAs.
        AU a[4];
#pragma unroll
        for (int mt = 0; mt < 4; ++mt) {
            const int rr = wm * 64 + mt * 16 + lane16;
            a[mt].h8[0] = *(const v8h*)&ldsx[buf][rr * LDS_STRIDE + lhalf * 16];
            a[mt].h8[1] = *(const v8h*)&ldsx[buf][rr * LDS_STRIDE + lhalf * 16 + 8];
        }
#pragma unroll
        for (int mt = 0; mt < 4; ++mt) {
#pragma unroll
            for (int nt = 0; nt < 2; ++nt) {
                acc[mt][nt] = __builtin_amdgcn_wmma_f32_16x16x32_f16(
                    false, a[mt].v, false, b[nt].v, (short)0, acc[mt][nt],
                    false, false);
            }
        }

        // Next tile must have landed before anyone reads it after the barrier.
        wait_async0();
        __syncthreads();
    }

    // Epilogue: C layout — VGPR r holds M=r (lanes 0-15) / M=r+8 (lanes 16-31).
#pragma unroll
    for (int mt = 0; mt < 4; ++mt) {
#pragma unroll
        for (int nt = 0; nt < 2; ++nt) {
#pragma unroll
            for (int r = 0; r < 8; ++r) {
                int rowg = mbase + wm * 64 + mt * 16 + lhalf * 8 + r;
                out[(size_t)rowg * N + gcol[nt]] = acc[mt][nt][r];
            }
        }
    }
}

extern "C" void kernel_launch(void* const* d_in, const int* in_sizes, int n_in,
                              void* d_out, int out_size, void* d_ws, size_t ws_size,
                              hipStream_t stream) {
    const _Float16* x       = (const _Float16*)d_in[0];
    const int*      qweight = (const int*)d_in[1];
    const int*      qzeros  = (const int*)d_in[2];
    const _Float16* scales  = (const _Float16*)d_in[3];
    float*          out     = (float*)d_out;

    const int  N = in_sizes[3];                          // 11008
    const long K = (long)in_sizes[1] * 8 / N;            // 4096
    const long M = (long)in_sizes[0] / K;                // 4096

    dim3 grid((unsigned)(N / BN), (unsigned)(M / BM));
    dim3 block(256);
    if (N == 11008 && K == 4096) {
        gptq4_wmma_gemm<11008, 4096><<<grid, block, 0, stream>>>(
            x, qweight, qzeros, scales, out, (int)N, (int)K);
    } else {
        gptq4_wmma_gemm<0, 0><<<grid, block, 0, stream>>>(
            x, qweight, qzeros, scales, out, (int)N, (int)K);
    }
}
